// Gemma3nTextAttention_2783138808561
// MI455X (gfx1250) — matrix operand order repaired and checked by reading, not compiled, not run on hardware
//
#include <hip/hip_runtime.h>
#include <hip/hip_bf16.h>

namespace {

constexpr int kB   = 2;
constexpr int kS   = 2048;
constexpr int kHid = 2048;
constexpr int kNH  = 8;
constexpr int kNKV = 2;
constexpr int kHD  = 256;
constexpr float kEps = 1e-6f;

typedef __attribute__((ext_vector_type(16))) __bf16 v16bf;
typedef __attribute__((ext_vector_type(8)))  float  v8f;

union Frag { v16bf v; unsigned int u[8]; };

__device__ __forceinline__ unsigned short f2bf(float f) {
  unsigned int u = __float_as_uint(f);
  u += 0x7FFFu + ((u >> 16) & 1u);
  return (unsigned short)(u >> 16);
}

__device__ __forceinline__ v8f wmma_bf16(const Frag& a, const Frag& b, v8f c) {
  return __builtin_amdgcn_wmma_f32_16x16x32_bf16(false, a.v, false, b.v,
                                                 (short)0, c, false, false);
}

__device__ __forceinline__ void wait_async0() {
  asm volatile("s_wait_asynccnt 0" ::: "memory");
}
__device__ __forceinline__ unsigned lds_off(const void* p) {
  return (unsigned)(unsigned long long)p;  // low 32 bits of LDS flat addr
}

// Async memory->LDS copy of 64B/lane: INST_OFFSET applies to both the global
// and the LDS address, so one address pair + literal offsets covers 4 chunks.
__device__ __forceinline__ void async_copy64(unsigned lds, const void* g) {
  asm volatile("global_load_async_to_lds_b128 %0, %1, off"
               :: "v"(lds), "v"(g) : "memory");
  asm volatile("global_load_async_to_lds_b128 %0, %1, off offset:16"
               :: "v"(lds), "v"(g) : "memory");
  asm volatile("global_load_async_to_lds_b128 %0, %1, off offset:32"
               :: "v"(lds), "v"(g) : "memory");
  asm volatile("global_load_async_to_lds_b128 %0, %1, off offset:48"
               :: "v"(lds), "v"(g) : "memory");
}

// interleave two bf16 streams: per-d (lo=even key, hi=odd key) u32 pairs
__device__ __forceinline__ void ilv(unsigned a, unsigned b,
                                    unsigned& lo, unsigned& hi) {
  lo = (a & 0xFFFFu) | (b << 16);
  hi = (a >> 16) | (b & 0xFFFF0000u);
}

// ---------- fp32 -> bf16 (RNE), 8 elements / thread ----------
__global__ __launch_bounds__(256) void cvt_bf16_kernel(
    const float* __restrict__ src, unsigned short* __restrict__ dst, long n) {
  long i = ((long)blockIdx.x * blockDim.x + threadIdx.x) * 8;
  if (i >= n) return;
  const float4 a = *(const float4*)(src + i);
  const float4 b = *(const float4*)(src + i + 4);
  uint4 o;
  o.x = (unsigned)f2bf(a.x) | ((unsigned)f2bf(a.y) << 16);
  o.y = (unsigned)f2bf(a.z) | ((unsigned)f2bf(a.w) << 16);
  o.z = (unsigned)f2bf(b.x) | ((unsigned)f2bf(b.y) << 16);
  o.w = (unsigned)f2bf(b.z) | ((unsigned)f2bf(b.w) << 16);
  *(uint4*)(dst + i) = o;
}

// ---------- C[M,N](f32) = A[M,K](bf16) * W[N,K]^T(bf16) ----------
// 128x128 block tile, BK=64, 8 waves: 4(m) x 2(n), wave patch 32x64.
// Double-buffered LDS; tiles staged with async memory->LDS copies that
// overlap the previous tile's WMMAs. One barrier per K-slice.
__global__ __launch_bounds__(256) void gemm_bf16_nt(
    const unsigned short* __restrict__ A, const unsigned short* __restrict__ W,
    float* __restrict__ C, int M, int N, int K) {
  __shared__ __align__(16) unsigned short As[2][128][64];  // 32 KB
  __shared__ __align__(16) unsigned short Bs[2][128][64];  // 32 KB
  const int tid = threadIdx.x;
  const int wid = tid >> 5, lane = tid & 31;
  const int wm = wid & 3, wn = wid >> 2;
  const int row0 = blockIdx.x * 128, col0 = blockIdx.y * 128;
  const int r = lane & 15, hi = lane >> 4;

  v8f acc[2][4];
  #pragma unroll
  for (int mt = 0; mt < 2; ++mt)
    #pragma unroll
    for (int nt = 0; nt < 4; ++nt)
      #pragma unroll
      for (int j = 0; j < 8; ++j) acc[mt][nt][j] = 0.0f;

  const int lr = tid >> 1, lc = (tid & 1) * 32;  // each thread: 64B row chunk
  const unsigned short* pa0 = A + (size_t)(row0 + lr) * K + lc;
  const unsigned short* pb0 = W + (size_t)(col0 + lr) * K + lc;

  // prologue: stage tile 0 into buffer 0
  async_copy64(lds_off(&As[0][lr][lc]), pa0);
  async_copy64(lds_off(&Bs[0][lr][lc]), pb0);

  const int nIter = K >> 6;
  for (int i = 0; i < nIter; ++i) {
    wait_async0();       // this wave's copies for tile i are resident
    __syncthreads();     // all waves' copies resident; prev reads retired
    const int cur = i & 1;
    if (i + 1 < nIter) { // stage tile i+1 under tile i's WMMAs
      const int kt2 = (i + 1) << 6;
      async_copy64(lds_off(&As[cur ^ 1][lr][lc]), pa0 + kt2);
      async_copy64(lds_off(&Bs[cur ^ 1][lr][lc]), pb0 + kt2);
    }
    #pragma unroll
    for (int ks = 0; ks < 2; ++ks) {
      Frag a[2];
      #pragma unroll
      for (int mt = 0; mt < 2; ++mt) {
        const int ar = wm * 32 + mt * 16 + r;
        #pragma unroll
        for (int j = 0; j < 4; ++j)
          a[mt].u[j] =
              *(const unsigned int*)&As[cur][ar][32 * ks + 8 * hi + 2 * j];
        #pragma unroll
        for (int j = 0; j < 4; ++j)
          a[mt].u[4 + j] =
              *(const unsigned int*)&As[cur][ar][32 * ks + 16 + 8 * hi + 2 * j];
      }
      #pragma unroll
      for (int nt = 0; nt < 4; ++nt) {
        Frag b;
        const int bc = wn * 64 + nt * 16 + r;
        #pragma unroll
        for (int j = 0; j < 8; ++j)
          b.u[j] =
              *(const unsigned int*)&Bs[cur][bc][32 * ks + 8 * hi + 2 * j + 8 * (j >> 2)];
        #pragma unroll
        for (int mt = 0; mt < 2; ++mt)
          acc[mt][nt] = wmma_bf16(a[mt], b, acc[mt][nt]);
      }
    }
  }

  #pragma unroll
  for (int mt = 0; mt < 2; ++mt)
    #pragma unroll
    for (int nt = 0; nt < 4; ++nt) {
      const int col = col0 + wn * 64 + nt * 16 + r;
      #pragma unroll
      for (int j = 0; j < 8; ++j) {
        const int row = row0 + wm * 32 + mt * 16 + j + 8 * hi;
        C[(size_t)row * N + col] = acc[mt][nt][j];
      }
    }
}

// ---------- RMSNorm (+ optional RoPE) over HD=256, one wave per vector ----
// src: (B*S, H, HD) f32;  dst: (B, H, S, HD) bf16
__global__ __launch_bounds__(256) void rmsnorm_rope_kernel(
    const float* __restrict__ src, unsigned short* __restrict__ dst,
    const float* __restrict__ w, const float* __restrict__ cosp,
    const float* __restrict__ sinp, int n_heads, int apply_rope) {
  const int gw = (blockIdx.x * blockDim.x + threadIdx.x) >> 5;
  const int lane = threadIdx.x & 31;
  if (gw >= kB * kS * n_heads) return;
  const int h = gw % n_heads;
  const int bs = gw / n_heads;
  const int s = bs % kS, b = bs / kS;
  const float* x = src + ((size_t)bs * n_heads + h) * kHD;
  const int d0 = lane * 8;  // lanes 0-15: d 0..127, lanes 16-31: d 128..255
  float v[8], ss = 0.0f;
  #pragma unroll
  for (int j = 0; j < 8; ++j) { v[j] = x[d0 + j]; ss += v[j] * v[j]; }
  #pragma unroll
  for (int m = 16; m >= 1; m >>= 1) ss += __shfl_xor(ss, m, 32);
  const float inv = rsqrtf(ss * (1.0f / kHD) + kEps);
  float y[8];
  #pragma unroll
  for (int j = 0; j < 8; ++j) y[j] = v[j] * inv * (w ? w[d0 + j] : 1.0f);
  if (apply_rope) {
    const float* cp = cosp + ((size_t)b * kS + s) * kHD + d0;
    const float* sp = sinp + ((size_t)b * kS + s) * kHD + d0;
    #pragma unroll
    for (int j = 0; j < 8; ++j) {
      const float other = __shfl_xor(y[j], 16, 32);  // rotate-half partner
      const float rot = (lane < 16) ? -other : other;
      y[j] = y[j] * cp[j] + rot * sp[j];
    }
  }
  unsigned short* o = dst + (((size_t)b * n_heads + h) * kS + s) * kHD + d0;
  #pragma unroll
  for (int j = 0; j < 8; j += 2)
    *(unsigned int*)(o + j) =
        (unsigned)f2bf(y[j]) | ((unsigned)f2bf(y[j + 1]) << 16);
}

// ---------- Flash attention: block = (128 q rows, one (b,h)) ----------
__global__ __launch_bounds__(256) void attn_kernel(
    const unsigned short* __restrict__ Qb,   // (B, NH, S, HD)
    const unsigned short* __restrict__ Kb,   // (B, NKV, S, HD)
    const unsigned short* __restrict__ Vb,   // (B, NKV, S, HD)
    unsigned short* __restrict__ Ob) {       // (B, S, NH*HD)
  __shared__ __align__(16) unsigned short Ks[32][kHD];   // 16 KB, row-major
  __shared__ __align__(16) unsigned int Vq[16][kHD];     // 16 KB, key-pair u32
  __shared__ __align__(16) unsigned short Ps[8][16][32]; // per-wave P patch
  const int bh = blockIdx.y;
  const int b = bh / kNH, h = bh % kNH, hk = h / (kNH / kNKV);
  const int qblk = blockIdx.x;
  const int tid = threadIdx.x, wid = tid >> 5, lane = tid & 31;
  const int r = lane & 15, hi = lane >> 4;
  const int q0 = qblk * 128 + wid * 16;

  const unsigned short* Qp = Qb + ((size_t)b * kNH + h) * kS * kHD;
  const unsigned short* Kp = Kb + ((size_t)b * kNKV + hk) * kS * kHD;
  const unsigned short* Vp = Vb + ((size_t)b * kNKV + hk) * kS * kHD;

  // Q fragments: 8 K-slices of 32, loaded once per wave tile
  Frag qf[8];
  #pragma unroll
  for (int t = 0; t < 8; ++t) {
    const unsigned short* qr = Qp + (size_t)(q0 + r) * kHD + 32 * t;
    #pragma unroll
    for (int j = 0; j < 4; ++j)
      qf[t].u[j] = *(const unsigned int*)(qr + 8 * hi + 2 * j);
    #pragma unroll
    for (int j = 0; j < 4; ++j)
      qf[t].u[4 + j] = *(const unsigned int*)(qr + 16 + 8 * hi + 2 * j);
  }

  v8f o[16];
  #pragma unroll
  for (int dt = 0; dt < 16; ++dt)
    #pragma unroll
    for (int j = 0; j < 8; ++j) o[dt][j] = 0.0f;
  float rm[8], rs[8];
  #pragma unroll
  for (int j = 0; j < 8; ++j) { rm[j] = -3.0e38f; rs[j] = 0.0f; }

  const int kv_end = (qblk + 1) * 128;        // causal: keys <= last q row
  const int klr = tid >> 3, klc = (tid & 7) * 32;   // K stage mapping
  const int vkp = tid >> 4, vd0 = (tid & 15) * 16;  // V stage mapping
  for (int k0 = 0; k0 < kv_end; k0 += 32) {
    // K tile: async memory -> LDS, row-major
    async_copy64(lds_off(&Ks[klr][klc]),
                 Kp + (size_t)(k0 + klr) * kHD + klc);
    // V tile: load two adjacent key rows, interleave per-d into u32 pairs
    {
      const uint4* a4 = (const uint4*)(Vp + (size_t)(k0 + 2 * vkp) * kHD + vd0);
      const uint4* b4 =
          (const uint4*)(Vp + (size_t)(k0 + 2 * vkp + 1) * kHD + vd0);
      const uint4 A0 = a4[0], A1 = a4[1];
      const uint4 B0 = b4[0], B1 = b4[1];
      unsigned w0[16];
      ilv(A0.x, B0.x, w0[0], w0[1]);   ilv(A0.y, B0.y, w0[2], w0[3]);
      ilv(A0.z, B0.z, w0[4], w0[5]);   ilv(A0.w, B0.w, w0[6], w0[7]);
      ilv(A1.x, B1.x, w0[8], w0[9]);   ilv(A1.y, B1.y, w0[10], w0[11]);
      ilv(A1.z, B1.z, w0[12], w0[13]); ilv(A1.w, B1.w, w0[14], w0[15]);
      uint4* dq = (uint4*)&Vq[vkp][vd0];
      #pragma unroll
      for (int c = 0; c < 4; ++c) {
        uint4 t;
        t.x = w0[4 * c]; t.y = w0[4 * c + 1];
        t.z = w0[4 * c + 2]; t.w = w0[4 * c + 3];
        dq[c] = t;
      }
      if (k0 + 32 < kv_end)  // warm L2 for the next V tile
        __builtin_prefetch(Vp + (size_t)(k0 + 32 + 2 * vkp) * kHD + vd0, 0, 0);
    }
    wait_async0();
    __syncthreads();

    // scores: two 16x16 tiles, K-dim = 256 -> 8 WMMA each
    v8f sc[2];
    #pragma unroll
    for (int nt = 0; nt < 2; ++nt) {
      #pragma unroll
      for (int j = 0; j < 8; ++j) sc[nt][j] = 0.0f;
      #pragma unroll
      for (int t = 0; t < 8; ++t) {
        Frag kf;
        const int key = nt * 16 + r;
        #pragma unroll
        for (int j = 0; j < 8; ++j)
          kf.u[j] = *(const unsigned int*)&Ks[key][32 * t + 8 * hi + 2 * j + 8 * (j >> 2)];
        sc[nt] = wmma_bf16(qf[t], kf, sc[nt]);
      }
    }
    // causal mask, only near the diagonal (wave-uniform branch)
    if (k0 + 31 > q0) {
      #pragma unroll
      for (int nt = 0; nt < 2; ++nt)
        #pragma unroll
        for (int j = 0; j < 8; ++j)
          if (k0 + nt * 16 + r > q0 + j + 8 * hi) sc[nt][j] = -1.0e9f;
    }

    // online softmax (row = j + 8*hi lives across the lane's 16-lane half)
    float nm[8];
    #pragma unroll
    for (int j = 0; j < 8; ++j) nm[j] = fmaxf(sc[0][j], sc[1][j]);
    #pragma unroll
    for (int m = 8; m >= 1; m >>= 1)
      #pragma unroll
      for (int j = 0; j < 8; ++j)
        nm[j] = fmaxf(nm[j], __shfl_xor(nm[j], m, 32));
    float scale[8], psum[8];
    #pragma unroll
    for (int j = 0; j < 8; ++j) {
      nm[j] = fmaxf(nm[j], rm[j]);
      scale[j] = __expf(rm[j] - nm[j]);
      rm[j] = nm[j];
      psum[j] = 0.0f;
    }
    #pragma unroll
    for (int nt = 0; nt < 2; ++nt)
      #pragma unroll
      for (int j = 0; j < 8; ++j) {
        const float p = __expf(sc[nt][j] - nm[j]);
        sc[nt][j] = p;
        psum[j] += p;
      }
    #pragma unroll
    for (int m = 8; m >= 1; m >>= 1)
      #pragma unroll
      for (int j = 0; j < 8; ++j) psum[j] += __shfl_xor(psum[j], m, 32);
    #pragma unroll
    for (int j = 0; j < 8; ++j) rs[j] = rs[j] * scale[j] + psum[j];
    #pragma unroll
    for (int dt = 0; dt < 16; ++dt)
      #pragma unroll
      for (int j = 0; j < 8; ++j) o[dt][j] *= scale[j];

    // re-shape P: C-layout -> A-layout via per-wave LDS patch
    #pragma unroll
    for (int nt = 0; nt < 2; ++nt)
      #pragma unroll
      for (int j = 0; j < 8; ++j)
        Ps[wid][j + 8 * hi][nt * 16 + r] = f2bf(sc[nt][j]);
    asm volatile("s_wait_dscnt 0" ::: "memory");
    __builtin_amdgcn_wave_barrier();
    Frag pf;
    #pragma unroll
    for (int j = 0; j < 4; ++j)
      pf.u[j] = *(const unsigned int*)&Ps[wid][r][8 * hi + 2 * j];
    #pragma unroll
    for (int j = 0; j < 4; ++j)
      pf.u[4 + j] = *(const unsigned int*)&Ps[wid][r][16 + 8 * hi + 2 * j];

    // O += P (16x32) * V (32x256): 16 WMMA across head dim
    #pragma unroll
    for (int dt = 0; dt < 16; ++dt) {
      Frag vf;
      const int d = dt * 16 + r;
      #pragma unroll
      for (int j = 0; j < 8; ++j) vf.u[j] = Vq[4 * hi + j + 4 * (j >> 2)][d];
      o[dt] = wmma_bf16(pf, vf, o[dt]);
    }
    __syncthreads();
  }

  // epilogue: normalize (multiply by 1/rs), store bf16 into (B, S, NH*HD)
  float irs[8];
  #pragma unroll
  for (int j = 0; j < 8; ++j) irs[j] = 1.0f / rs[j];
  #pragma unroll
  for (int dt = 0; dt < 16; ++dt) {
    const int d = dt * 16 + r;
    #pragma unroll
    for (int j = 0; j < 8; ++j) {
      const int row = q0 + j + 8 * hi;
      Ob[((size_t)b * kS + row) * (kNH * kHD) + (size_t)h * kHD + d] =
          f2bf(o[dt][j] * irs[j]);
    }
  }
}

}  // namespace

extern "C" void kernel_launch(void* const* d_in, const int* in_sizes, int n_in,
                              void* d_out, int out_size, void* d_ws,
                              size_t ws_size, hipStream_t stream) {
  const float* hs   = (const float*)d_in[0];
  const float* cosp = (const float*)d_in[1];
  const float* sinp = (const float*)d_in[2];
  // d_in[3] attention_mask: pure causal, applied analytically in-kernel
  const float* Wq = (const float*)d_in[4];
  const float* Wk = (const float*)d_in[5];
  const float* Wv = (const float*)d_in[6];
  const float* Wo = (const float*)d_in[7];
  const float* qw = (const float*)d_in[8];
  const float* kw = (const float*)d_in[9];
  float* out = (float*)d_out;

  char* ws = (char*)d_ws;
  const size_t M    = (size_t)kB * kS;            // 4096
  const size_t nHS  = M * kHid;
  const size_t nWq  = (size_t)kNH * kHD * kHid;   // 2048x2048
  const size_t nWkv = (size_t)kNKV * kHD * kHid;  // 512x2048
  const size_t nQ   = M * kNH * kHD;
  const size_t nKV  = M * kNKV * kHD;

  size_t off = 0;
  unsigned short* hsb = (unsigned short*)(ws + off); off += nHS * 2;
  unsigned short* Wqb = (unsigned short*)(ws + off); off += nWq * 2;
  unsigned short* Wkb = (unsigned short*)(ws + off); off += nWkv * 2;
  unsigned short* Wvb = (unsigned short*)(ws + off); off += nWkv * 2;
  unsigned short* Wob = (unsigned short*)(ws + off); off += nWq * 2;
  unsigned short* qb  = (unsigned short*)(ws + off); off += nQ * 2;
  unsigned short* kb  = (unsigned short*)(ws + off); off += nKV * 2;
  unsigned short* vb  = (unsigned short*)(ws + off); off += nKV * 2;
  unsigned short* ab  = (unsigned short*)(ws + off); off += nQ * 2;
  float* qf32 = (float*)(ws + off); off += nQ * 4;
  float* kf32 = (float*)(ws + off); off += nKV * 4;
  float* vf32 = (float*)(ws + off); off += nKV * 4;

  auto cvt = [&](const float* s, unsigned short* d, size_t n) {
    cvt_bf16_kernel<<<dim3((unsigned)((n / 8 + 255) / 256)), 256, 0, stream>>>(
        s, d, (long)n);
  };
  cvt(hs, hsb, nHS);
  cvt(Wq, Wqb, nWq);
  cvt(Wk, Wkb, nWkv);
  cvt(Wv, Wvb, nWkv);
  cvt(Wo, Wob, nWq);

  gemm_bf16_nt<<<dim3(M / 128, (kNH * kHD) / 128), 256, 0, stream>>>(
      hsb, Wqb, qf32, (int)M, kNH * kHD, kHid);
  gemm_bf16_nt<<<dim3(M / 128, (kNKV * kHD) / 128), 256, 0, stream>>>(
      hsb, Wkb, kf32, (int)M, kNKV * kHD, kHid);
  gemm_bf16_nt<<<dim3(M / 128, (kNKV * kHD) / 128), 256, 0, stream>>>(
      hsb, Wvb, vf32, (int)M, kNKV * kHD, kHid);

  rmsnorm_rope_kernel<<<(kB * kS * kNH) / 8, 256, 0, stream>>>(
      qf32, qb, qw, cosp, sinp, kNH, 1);
  rmsnorm_rope_kernel<<<(kB * kS * kNKV) / 8, 256, 0, stream>>>(
      kf32, kb, kw, cosp, sinp, kNKV, 1);
  rmsnorm_rope_kernel<<<(kB * kS * kNKV) / 8, 256, 0, stream>>>(
      vf32, vb, nullptr, cosp, sinp, kNKV, 0);

  attn_kernel<<<dim3(kS / 128, kB * kNH), 256, 0, stream>>>(qb, kb, vb, ab);

  gemm_bf16_nt<<<dim3(M / 128, kHid / 128), 256, 0, stream>>>(
      ab, Wob, out, (int)M, kHid, kHid);

  (void)in_sizes; (void)n_in; (void)out_size; (void)ws_size; (void)off;
}
